// LSTM_21869973471591
// MI455X (gfx1250) — matrix operand order repaired and checked by reading, not compiled, not run on hardware
//
#include <hip/hip_runtime.h>

typedef __attribute__((ext_vector_type(16))) _Float16 v16h;
typedef __attribute__((ext_vector_type(8)))  _Float16 v8h;
typedef __attribute__((ext_vector_type(4)))  _Float16 v4h;
typedef __attribute__((ext_vector_type(8)))  float    v8f;
typedef __attribute__((ext_vector_type(4)))  float    v4f;

#define TSEQ 200
#define HID  64
#define HROW 104   // padded row stride (halves): 52 dwords -> conflict-free b128 A fetch

// ---- fast activations ----
// i/f/o gate accumulators are pre-scaled by 0.5 via B, so sigmoid needs no input mul.
#if defined(__AMDGCN__) && __has_builtin(__builtin_amdgcn_tanhf)
__device__ __forceinline__ float ftanh(float x)      { return __builtin_amdgcn_tanhf(x); }
__device__ __forceinline__ float fsig_pre(float xh)  { return fmaf(0.5f, __builtin_amdgcn_tanhf(xh), 0.5f); }
#else
#define LOG2E 1.4426950408889634f
__device__ __forceinline__ float ftanh(float x) {
    return 1.0f - 2.0f * __builtin_amdgcn_rcpf(__builtin_amdgcn_exp2f(2.0f * LOG2E * x) + 1.0f);
}
__device__ __forceinline__ float fsig_pre(float xh) {   // xh = x/2
    return __builtin_amdgcn_rcpf(1.0f + __builtin_amdgcn_exp2f(-2.0f * LOG2E * xh));
}
#endif

__global__ __launch_bounds__(256, 1)
void lstm_wmma_kernel(const float* __restrict__ x,
                      const float* __restrict__ W_ih,
                      const float* __restrict__ W_hh,
                      const float* __restrict__ b_ih,
                      const float* __restrict__ b_hh,
                      const float* __restrict__ W_fc,
                      const float* __restrict__ b_fc,
                      float* __restrict__ out)
{
    // Two independent 16-row batch tiles per workgroup (2 waves/SIMD for latency hiding).
    __shared__ _Float16 x_lds[2][16 * TSEQ * 2];   // staged x, f16 (25.6 KB)
    __shared__ _Float16 h_lds[2][2][16 * HROW];    // per-tile double-buffered [h|x|1|0..] rows

    const int tid  = threadIdx.x;
    const int ti   = tid >> 7;        // tile 0/1 within workgroup
    const int lt   = tid & 127;       // thread id within tile group
    const int wv   = lt >> 5;         // wave 0..3 within tile -> hidden cols 16*wv..
    const int lane = tid & 31;
    const int l16  = lane & 15;
    const int hl   = lane >> 4;       // half-wave select
    const int b0   = blockIdx.x * 32 + ti * 16;
    const int j    = wv * 16 + l16;   // hidden column owned in C/D layout

    // ---- stage x (32 contiguous rows = 12800 floats) into LDS as f16, linear copy ----
    {
        const v4f* xg = (const v4f*)(x + (size_t)blockIdx.x * 32 * TSEQ * 2);
        _Float16*  xs = &x_lds[0][0];
        for (int idx = tid; idx < 32 * TSEQ * 2 / 4; idx += 256) {
            v4f v = xg[idx];
            v4h h4;
            h4[0] = (_Float16)v[0]; h4[1] = (_Float16)v[1];
            h4[2] = (_Float16)v[2]; h4[3] = (_Float16)v[3];
            *(v4h*)&xs[idx * 4] = h4;
        }
    }
    // ---- zero all h buffers (h0 = 0, pad cols = 0) ----
    {
        _Float16* hz = &h_lds[0][0][0];
        for (int idx = tid; idx < 2 * 2 * 16 * HROW; idx += 256)
            hz[idx] = (_Float16)0.0f;
    }

    // ---- persistent B-matrix VGPRs (WMMA B layout), f16 ----
    // chunks 0,1: B[k][n] = s * W_hh[n][k]; chunk 2: k=64,65 -> s*W_ih[n][:], k=66 -> s*(b_ih+b_hh)
    // s = 0.5 for i/f/o gates (pre-scales sigmoid input), 1.0 for g gate.
    v16h Bm[4][3];
#pragma unroll
    for (int tt = 0; tt < 4; ++tt) {
        const int   n = tt * 64 + j;
        const float s = (tt == 2) ? 1.0f : 0.5f;
#pragma unroll
        for (int ch = 0; ch < 2; ++ch) {
            const float* src = W_hh + (size_t)n * HID + ch * 32 + hl * 16;
            v16h bv;
#pragma unroll
            for (int e = 0; e < 16; ++e) bv[e] = (_Float16)(s * src[e]);
            Bm[tt][ch] = bv;
        }
        v16h bv;
#pragma unroll
        for (int e = 0; e < 16; ++e) bv[e] = (_Float16)0.0f;
        if (hl == 0) {
            bv[0] = (_Float16)(s * W_ih[n * 2 + 0]);
            bv[1] = (_Float16)(s * W_ih[n * 2 + 1]);
            bv[2] = (_Float16)(s * (b_ih[n] + b_hh[n]));
        }
        Bm[tt][2] = bv;
    }

    float c[8];
#pragma unroll
    for (int r = 0; r < 8; ++r) c[r] = 0.0f;

    __syncthreads();   // zero-init + x staging visible

    // ---- seed augmented columns of buffer 0: x(t=0) and the constant-1 column ----
    if (tid < 64) {
        int t2 = tid >> 5, r = tid & 31;
        int m = r >> 1, w2 = r & 1;
        h_lds[t2][0][m * HROW + 64 + w2] = x_lds[t2][(m * TSEQ + 0) * 2 + w2];
    }
    if (tid < 32) {
        int t2 = tid >> 4, m = tid & 15;
        h_lds[t2][0][m * HROW + 66] = (_Float16)1.0f;
        h_lds[t2][1][m * HROW + 66] = (_Float16)1.0f;
    }
    __syncthreads();

    const v8f zacc = {};

    int cur = 0;
#pragma unroll 1
    for (int t = 0; t < TSEQ; ++t) {
        // ---- A fetch (16-bit A layout: lane row M = l16, contiguous k octets) ----
        v16h A[3];
#pragma unroll
        for (int ch = 0; ch < 3; ++ch) {
            v8h lo = *(const v8h*)&h_lds[ti][cur][l16 * HROW + ch * 32 + hl * 16];
            v8h hi = *(const v8h*)&h_lds[ti][cur][l16 * HROW + ch * 32 + hl * 16 + 8];
            v16h a;
#pragma unroll
            for (int e = 0; e < 8; ++e) { a[e] = lo[e]; a[8 + e] = hi[e]; }
            A[ch] = a;
        }

        // ---- gates = [h|x|1] @ [W_hh;W_ih;b]^T : 12x v_wmma_f32_16x16x32_f16 ----
        v8f acc[4];
#pragma unroll
        for (int tt = 0; tt < 4; ++tt) {
            acc[tt] = __builtin_amdgcn_wmma_f32_16x16x32_f16(
                false, A[0], false, Bm[tt][0], (short)0, zacc, false, false);
            acc[tt] = __builtin_amdgcn_wmma_f32_16x16x32_f16(
                false, A[1], false, Bm[tt][1], (short)0, acc[tt], false, false);
            acc[tt] = __builtin_amdgcn_wmma_f32_16x16x32_f16(
                false, A[2], false, Bm[tt][2], (short)0, acc[tt], false, false);
        }

        const int nxt = cur ^ 1;

        // ---- refresh augmented x columns of the next buffer (wave 0 of each tile) ----
        if (wv == 0 && t + 1 < TSEQ) {
            int m = lane >> 1, w2 = lane & 1;
            h_lds[ti][nxt][m * HROW + 64 + w2] = x_lds[ti][(m * TSEQ + (t + 1)) * 2 + w2];
        }

        // ---- elementwise LSTM cell update (lane owns col j, rows r+8*hl) ----
#pragma unroll
        for (int r = 0; r < 8; ++r) {
            float ig = fsig_pre(acc[0][r]);   // acc pre-scaled by 0.5 via B
            float fg = fsig_pre(acc[1][r]);
            float gg = ftanh(acc[2][r]);
            float og = fsig_pre(acc[3][r]);
            float cn = fmaf(fg, c[r], ig * gg);
            c[r]     = cn;
            float hn = og * ftanh(cn);
            int m    = r + hl * 8;
            h_lds[ti][nxt][m * HROW + j] = (_Float16)hn;
        }
        __syncthreads();
        cur = nxt;
    }

    // ---- out[b] = h_last @ W_fc.T + b_fc (read final f16 h from LDS) ----
    if (tid < 32) {
        int t2 = tid >> 4, m = tid & 15;
        float s = b_fc[0];
#pragma unroll 8
        for (int jj = 0; jj < HID; ++jj)
            s += (float)h_lds[t2][cur][m * HROW + jj] * W_fc[jj];
        out[blockIdx.x * 32 + t2 * 16 + m] = s;
    }
}

extern "C" void kernel_launch(void* const* d_in, const int* in_sizes, int n_in,
                              void* d_out, int out_size, void* d_ws, size_t ws_size,
                              hipStream_t stream) {
    const float* x    = (const float*)d_in[0];
    const float* W_ih = (const float*)d_in[1];
    const float* W_hh = (const float*)d_in[2];
    const float* b_ih = (const float*)d_in[3];
    const float* b_hh = (const float*)d_in[4];
    const float* W_fc = (const float*)d_in[5];
    const float* b_fc = (const float*)d_in[6];

    const int B      = in_sizes[0] / (TSEQ * 2); // 4096
    const int blocks = B / 32;                   // 128 workgroups, two 16-row tiles each

    lstm_wmma_kernel<<<blocks, 256, 0, stream>>>(x, W_ih, W_hh, b_ih, b_hh,
                                                 W_fc, b_fc, (float*)d_out);
}
